// MoE_26113401160074
// MI455X (gfx1250) — compile-verified, matrix operand order checked by reading, not run on hardware
//
#include <hip/hip_runtime.h>
#include <hip/hip_bf16.h>

// ---------------------------------------------------------------------------
// MoE top-2 FFN for MI455X (gfx1250, wave32, WMMA bf16).
//   router -> scan -> gathered GEMM1 (x@[W1|W3] fused + SiLU) -> GEMM2 (h@W2,
//   atomic scatter-add into out). Double-buffered LDS, packed bf16 staging.
// ---------------------------------------------------------------------------

#define NTOK   8192          // bs*slen = 4*2048
#define DIM    2048
#define HID    1024
#define NEXP   8

#define BM     128
#define BN     128
#define BK     32
#define LDK    (BK + 8)      // 40 elems = 80B row stride; keeps 16B alignment

typedef __attribute__((ext_vector_type(16))) __bf16 bf16x16;
typedef __attribute__((ext_vector_type(8)))  float  f32x8;

union FragU { uint4 u[2]; bf16x16 f; };
struct __align__(16) Pack16 { __bf16 v[16]; };

// Load a 16x32 bf16 fragment from LDS stored as [row/col][K], stride LDK.
// Lanes 0-15: K 0-7 / 16-23, lanes 16-31: K 8-15 / 24-31 (CDNA5 layout)
// => two 16-byte-aligned ds_load_b128 per fragment.
__device__ __forceinline__ bf16x16 frag_ld(const __bf16* s, int base, int lane) {
    const int i  = base + (lane & 15);
    const int kh = (lane >> 4) * 8;
    FragU fu;
    fu.u[0] = *reinterpret_cast<const uint4*>(s + i * LDK + kh);
    fu.u[1] = *reinterpret_cast<const uint4*>(s + i * LDK + 16 + kh);
    return fu.f;
}

__device__ __forceinline__ f32x8 wmma_bf16(bf16x16 a, bf16x16 b, f32x8 c) {
    return __builtin_amdgcn_wmma_f32_16x16x32_bf16(
        false, a, false, b, (short)0, c, false, false);
}

// Pack two fp32 -> one dword of {bf16(lo), bf16(hi)} and store to LDS [n][k]
// (k even => 4B aligned). Emits v_cvt_pk_bf16_f32 + ds_store_b32.
__device__ __forceinline__ void st_pk(__bf16* s, int n, int k, float lo, float hi) {
    union { __bf16 b[2]; unsigned int u; } pk;
    pk.b[0] = (__bf16)lo;
    pk.b[1] = (__bf16)hi;
    *reinterpret_cast<unsigned int*>(s + n * LDK + k) = pk.u;
}

// Stage a 32(K) x 128(N) fp32 weight tile transposed to LDS [n][k] bf16.
// Thread owns K-pair kp,kp+1 x 8 N values => 4 global_load_b128, 8 ds_store_b32.
__device__ __forceinline__ void stage_w(__bf16* sB, const float* w, int ldw,
                                        int kk, int n0, int kp, int nsg) {
    const float4* p0 = reinterpret_cast<const float4*>(w + (size_t)(kk + kp) * ldw + n0 + nsg);
    const float4* p1 = reinterpret_cast<const float4*>(w + (size_t)(kk + kp + 1) * ldw + n0 + nsg);
    const float4 a0 = p0[0], a1 = p0[1], b0 = p1[0], b1 = p1[1];
    st_pk(sB, nsg + 0, kp, a0.x, b0.x);
    st_pk(sB, nsg + 1, kp, a0.y, b0.y);
    st_pk(sB, nsg + 2, kp, a0.z, b0.z);
    st_pk(sB, nsg + 3, kp, a0.w, b0.w);
    st_pk(sB, nsg + 4, kp, a1.x, b1.x);
    st_pk(sB, nsg + 5, kp, a1.y, b1.y);
    st_pk(sB, nsg + 6, kp, a1.z, b1.z);
    st_pk(sB, nsg + 7, kp, a1.w, b1.w);
}

// ---------------------------------------------------------------------------
// Kernel 1: router. One wave32 per token.
// ---------------------------------------------------------------------------
__global__ __launch_bounds__(256) void moe_router(
    const float* __restrict__ x, const float* __restrict__ gw,
    int* __restrict__ cnt, int* __restrict__ idx, float* __restrict__ scl) {
    const int lane = threadIdx.x & 31;
    const int wid  = threadIdx.x >> 5;
    const int t    = blockIdx.x * 8 + wid;
    if (t >= NTOK) return;

    float a[NEXP];
#pragma unroll
    for (int e = 0; e < NEXP; ++e) a[e] = 0.f;

    const float* xr = x + (size_t)t * DIM;
    for (int k = lane; k < DIM; k += 32) {
        const float xv = xr[k];
        const float4* g = reinterpret_cast<const float4*>(gw + (size_t)k * NEXP);
        const float4 g0 = g[0], g1 = g[1];
        a[0] += xv * g0.x; a[1] += xv * g0.y; a[2] += xv * g0.z; a[3] += xv * g0.w;
        a[4] += xv * g1.x; a[5] += xv * g1.y; a[6] += xv * g1.z; a[7] += xv * g1.w;
    }
#pragma unroll
    for (int e = 0; e < NEXP; ++e)
#pragma unroll
        for (int off = 16; off > 0; off >>= 1)
            a[e] += __shfl_down(a[e], off, 32);

    if (lane == 0) {
        float mx = a[0];
#pragma unroll
        for (int e = 1; e < NEXP; ++e) mx = fmaxf(mx, a[e]);
        float p[NEXP], s = 0.f;
#pragma unroll
        for (int e = 0; e < NEXP; ++e) { p[e] = __expf(a[e] - mx); s += p[e]; }
        const float inv = 1.f / s;

        int i1 = 0; float v1 = p[0];
#pragma unroll
        for (int e = 1; e < NEXP; ++e) if (p[e] > v1) { v1 = p[e]; i1 = e; }
        int i2 = -1; float v2 = -1.f;
#pragma unroll
        for (int e = 0; e < NEXP; ++e)
            if (e != i1 && p[e] > v2) { v2 = p[e]; i2 = e; }

        int pos = atomicAdd(&cnt[i1], 1);
        idx[i1 * NTOK + pos] = t; scl[i1 * NTOK + pos] = v1 * inv;
        pos = atomicAdd(&cnt[i2], 1);
        idx[i2 * NTOK + pos] = t; scl[i2 * NTOK + pos] = v2 * inv;
    }
}

// ---------------------------------------------------------------------------
// Kernel 2: exclusive scan of expert counts (H row offsets).
// ---------------------------------------------------------------------------
__global__ void moe_scan(const int* __restrict__ cnt, int* __restrict__ offs) {
    if (threadIdx.x == 0) {
        int s = 0;
        for (int e = 0; e < NEXP; ++e) { offs[e] = s; s += cnt[e]; }
        offs[NEXP] = s;
    }
}

// ---------------------------------------------------------------------------
// Kernel 3: gathered GEMM1: h = silu((s*x)@W1_e) * ((s*x)@W3_e) -> bf16 H.
// Double-buffered LDS; one barrier per K-step.
// ---------------------------------------------------------------------------
__global__ __launch_bounds__(256) void moe_gemm1(
    const float* __restrict__ x,  const float* __restrict__ w1,
    const float* __restrict__ w3, const int* __restrict__ cnt,
    const int* __restrict__ offs, const int* __restrict__ idx,
    const float* __restrict__ scl, __bf16* __restrict__ H) {
    __shared__ __align__(16) __bf16 sA [2][BM * LDK];
    __shared__ __align__(16) __bf16 sB1[2][BN * LDK];
    __shared__ __align__(16) __bf16 sB3[2][BN * LDK];
    __shared__ int   sTok[BM];
    __shared__ float sScl[BM];

    const int e    = blockIdx.x >> 6;
    const int tm   = blockIdx.x & 63;
    const int ne   = cnt[e];
    const int row0 = tm * BM;
    if (row0 >= ne) return;
    const int hoff = offs[e];
    const int n0   = blockIdx.y * BN;

    const float* w1e = w1 + (size_t)e * DIM * HID;
    const float* w3e = w3 + (size_t)e * DIM * HID;

    const int tid = threadIdx.x;
    if (tid < BM) {
        const int gr = row0 + tid;
        if (gr < ne) { sTok[tid] = idx[e * NTOK + gr]; sScl[tid] = scl[e * NTOK + gr]; }
        else         { sTok[tid] = -1;                 sScl[tid] = 0.f; }
    }
    __syncthreads();

    const int lane = tid & 31, wid = tid >> 5;
    const int wm = (wid >> 2) * 64;   // 0 / 64
    const int wn = (wid & 3) * 32;    // 0,32,64,96

    const f32x8 z = {0.f,0.f,0.f,0.f,0.f,0.f,0.f,0.f};
    f32x8 acc1[4][2], acc3[4][2];
#pragma unroll
    for (int i = 0; i < 4; ++i)
#pragma unroll
        for (int j = 0; j < 2; ++j) { acc1[i][j] = z; acc3[i][j] = z; }

    // fixed staging coordinates
    const int arow = tid >> 1, ahalf = (tid & 1) * 16;
    const int tok  = sTok[arow];
    const float asc = sScl[arow];
    const float* xrow = x + (size_t)(tok >= 0 ? tok : 0) * DIM;
    const int kp  = (tid >> 4) * 2;       // K pair base: 0,2,..,30
    const int nsg = (tid & 15) * 8;       // N segment: 0,8,..,120

    // ---- staging helpers (inlined) -------------------------------------
    auto stage = [&](int buf, int kk) {
        // A: gathered, scaled, fp32 -> bf16, row-major [m][k]
        Pack16 pa;
        if (tok >= 0) {
            const float4* src = reinterpret_cast<const float4*>(xrow + kk + ahalf);
#pragma unroll
            for (int q = 0; q < 4; ++q) {
                const float4 v = src[q];
                pa.v[q*4+0] = (__bf16)(v.x * asc);
                pa.v[q*4+1] = (__bf16)(v.y * asc);
                pa.v[q*4+2] = (__bf16)(v.z * asc);
                pa.v[q*4+3] = (__bf16)(v.w * asc);
            }
        } else {
#pragma unroll
            for (int q = 0; q < 16; ++q) pa.v[q] = (__bf16)0.f;
        }
        *reinterpret_cast<Pack16*>(&sA[buf][arow * LDK + ahalf]) = pa;
        // B1/B3: transposed [n][k], packed bf16x2 stores
        stage_w(sB1[buf], w1e, HID, kk, n0, kp, nsg);
        stage_w(sB3[buf], w3e, HID, kk, n0, kp, nsg);
    };

    auto compute = [&](int buf) {
        bf16x16 bf1[2], bf3[2];
#pragma unroll
        for (int nt = 0; nt < 2; ++nt) {
            bf1[nt] = frag_ld(sB1[buf], wn + nt*16, lane);
            bf3[nt] = frag_ld(sB3[buf], wn + nt*16, lane);
        }
#pragma unroll
        for (int mt = 0; mt < 4; ++mt) {
            const bf16x16 af = frag_ld(sA[buf], wm + mt*16, lane);
#pragma unroll
            for (int nt = 0; nt < 2; ++nt) {
                acc1[mt][nt] = wmma_bf16(af, bf1[nt], acc1[mt][nt]);
                acc3[mt][nt] = wmma_bf16(af, bf3[nt], acc3[mt][nt]);
            }
        }
    };

    constexpr int ITER = DIM / BK;   // 64
    stage(0, 0);
    for (int i = 0; i < ITER; ++i) {
        __syncthreads();
        const int buf = i & 1;
        if (i + 1 < ITER) {
            if (i + 2 < ITER)   // hint tile-after-next into GL2
                __builtin_prefetch(w1e + (size_t)((i + 2) * BK + kp) * HID + n0 + nsg, 0, 1);
            stage(buf ^ 1, (i + 1) * BK);
        }
        compute(buf);
    }

    // ---- epilogue: h = silu(c1) * c3 -> bf16 H
    const int hi = lane >> 4, nn = lane & 15;
#pragma unroll
    for (int mt = 0; mt < 4; ++mt)
#pragma unroll
        for (int nt = 0; nt < 2; ++nt)
#pragma unroll
            for (int r = 0; r < 8; ++r) {
                const int rloc = wm + mt*16 + r + 8*hi;
                const int gr   = row0 + rloc;
                if (gr < ne) {
                    const int gc = n0 + wn + nt*16 + nn;
                    const float c1 = acc1[mt][nt][r];
                    const float c3 = acc3[mt][nt][r];
                    const float hv = c1 * (1.f / (1.f + __expf(-c1))) * c3;
                    H[(size_t)(hoff + gr) * HID + gc] = (__bf16)hv;
                }
            }
}

// ---------------------------------------------------------------------------
// Kernel 4: GEMM2. out[token] += H_e @ W2_e, f32 atomic scatter-add.
// ---------------------------------------------------------------------------
__global__ __launch_bounds__(256) void moe_gemm2(
    const __bf16* __restrict__ H, const float* __restrict__ w2,
    const int* __restrict__ cnt,  const int* __restrict__ offs,
    const int* __restrict__ idx,  float* __restrict__ out) {
    __shared__ __align__(16) __bf16 sA[2][BM * LDK];
    __shared__ __align__(16) __bf16 sB[2][BN * LDK];
    __shared__ int sTok[BM];

    const int e    = blockIdx.x >> 6;
    const int tm   = blockIdx.x & 63;
    const int ne   = cnt[e];
    const int row0 = tm * BM;
    if (row0 >= ne) return;
    const int hoff = offs[e];
    const int n0   = blockIdx.y * BN;
    const float* w2e = w2 + (size_t)e * HID * DIM;

    const int tid = threadIdx.x;
    if (tid < BM) {
        const int gr = row0 + tid;
        sTok[tid] = (gr < ne) ? idx[e * NTOK + gr] : -1;
    }
    __syncthreads();

    const int lane = tid & 31, wid = tid >> 5;
    const int wm = (wid >> 2) * 64;
    const int wn = (wid & 3) * 32;

    const f32x8 z = {0.f,0.f,0.f,0.f,0.f,0.f,0.f,0.f};
    f32x8 acc[4][2];
#pragma unroll
    for (int i = 0; i < 4; ++i)
#pragma unroll
        for (int j = 0; j < 2; ++j) acc[i][j] = z;

    const int arow = tid >> 1, ahalf = (tid & 1) * 16;
    const bool avalid = (row0 + arow) < ne;
    const __bf16* hrow = H + (size_t)(hoff + row0 + (avalid ? arow : 0)) * HID;
    const int kp  = (tid >> 4) * 2;
    const int nsg = (tid & 15) * 8;

    auto stage = [&](int buf, int kk) {
        if (avalid) {
            const uint4* src = reinterpret_cast<const uint4*>(hrow + kk + ahalf);
            uint4* dst = reinterpret_cast<uint4*>(&sA[buf][arow * LDK + ahalf]);
            dst[0] = src[0];
            dst[1] = src[1];
        } else {
            Pack16 pz;
#pragma unroll
            for (int q = 0; q < 16; ++q) pz.v[q] = (__bf16)0.f;
            *reinterpret_cast<Pack16*>(&sA[buf][arow * LDK + ahalf]) = pz;
        }
        stage_w(sB[buf], w2e, DIM, kk, n0, kp, nsg);
    };

    auto compute = [&](int buf) {
        bf16x16 bf[2];
#pragma unroll
        for (int nt = 0; nt < 2; ++nt) bf[nt] = frag_ld(sB[buf], wn + nt*16, lane);
#pragma unroll
        for (int mt = 0; mt < 4; ++mt) {
            const bf16x16 af = frag_ld(sA[buf], wm + mt*16, lane);
#pragma unroll
            for (int nt = 0; nt < 2; ++nt)
                acc[mt][nt] = wmma_bf16(af, bf[nt], acc[mt][nt]);
        }
    };

    constexpr int ITER = HID / BK;   // 32
    stage(0, 0);
    for (int i = 0; i < ITER; ++i) {
        __syncthreads();
        const int buf = i & 1;
        if (i + 1 < ITER) {
            if (i + 2 < ITER)
                __builtin_prefetch(w2e + (size_t)((i + 2) * BK + kp) * DIM + n0 + nsg, 0, 1);
            stage(buf ^ 1, (i + 1) * BK);
        }
        compute(buf);
    }

    // ---- epilogue: atomic scatter-add (each token owned by exactly 2 experts)
    const int hi = lane >> 4, nn = lane & 15;
#pragma unroll
    for (int mt = 0; mt < 4; ++mt)
#pragma unroll
        for (int nt = 0; nt < 2; ++nt)
#pragma unroll
            for (int r = 0; r < 8; ++r) {
                const int rloc = wm + mt*16 + r + 8*hi;
                if (row0 + rloc < ne) {
                    const int tkn = sTok[rloc];
                    const int gc  = n0 + wn + nt*16 + nn;
                    atomicAdd(&out[(size_t)tkn * DIM + gc], acc[mt][nt][r]);
                }
            }
}

// ---------------------------------------------------------------------------
// Host-side launcher (graph-capture safe: async memsets + kernels only).
// ---------------------------------------------------------------------------
extern "C" void kernel_launch(void* const* d_in, const int* in_sizes, int n_in,
                              void* d_out, int out_size, void* d_ws, size_t ws_size,
                              hipStream_t stream) {
    const float* x    = (const float*)d_in[0];
    const float* gate = (const float*)d_in[1];
    const float* w1   = (const float*)d_in[2];
    const float* w2   = (const float*)d_in[3];
    const float* w3   = (const float*)d_in[4];
    float* out = (float*)d_out;

    char* ws = (char*)d_ws;
    const size_t OFF_IDX = 4096;
    const size_t OFF_SCL = OFF_IDX + (size_t)NEXP * NTOK * sizeof(int);
    const size_t OFF_H   = OFF_SCL + (size_t)NEXP * NTOK * sizeof(float);

    int*    cnt  = (int*)(ws);
    int*    offs = (int*)(ws + 64);
    int*    idx  = (int*)(ws + OFF_IDX);
    float*  scl  = (float*)(ws + OFF_SCL);
    __bf16* H    = (__bf16*)(ws + OFF_H);

    hipMemsetAsync(ws, 0, 4096, stream);                                // counters
    hipMemsetAsync(d_out, 0, (size_t)out_size * sizeof(float), stream); // accum dst

    moe_router<<<NTOK / 8, 256, 0, stream>>>(x, gate, cnt, idx, scl);
    moe_scan<<<1, 32, 0, stream>>>(cnt, offs);

    dim3 g1(NEXP * (NTOK / BM), HID / BN);   // 512 x 8
    moe_gemm1<<<g1, 256, 0, stream>>>(x, w1, w3, cnt, offs, idx, scl, H);

    dim3 g2(NEXP * (NTOK / BM), DIM / BN);   // 512 x 16
    moe_gemm2<<<g2, 256, 0, stream>>>(H, w2, cnt, offs, idx, out);
}
